// Decoder_23708219474393
// MI455X (gfx1250) — matrix operand order repaired and checked by reading, not compiled, not run on hardware
//
#include <hip/hip_runtime.h>
#include <math.h>

// Problem constants
#define BB 32
#define TT 512
#define VV 1024
#define HH 1024
#define G4 4096   // 4*H

typedef __attribute__((ext_vector_type(16))) __bf16 v16bf;
typedef __attribute__((ext_vector_type(8)))  float  v8f;
typedef int i32x4 __attribute__((vector_size(16)));   // matches builtin pointee

union Frag16 { v16bf v; unsigned u[8]; };

// --- CDNA5 async global->LDS copy (VGPR-bypassing, ASYNCcnt-tracked) -------
#ifdef __has_builtin
#  if __has_builtin(__builtin_amdgcn_global_load_async_to_lds_b128)
#    define USE_ASYNC_LDS 1
#  endif
#endif
#ifndef USE_ASYNC_LDS
#  define USE_ASYNC_LDS 0
#endif

__device__ __forceinline__ void cp16(const unsigned short* g, unsigned short* l) {
#if USE_ASYNC_LDS
    __builtin_amdgcn_global_load_async_to_lds_b128(
        (__attribute__((address_space(1))) i32x4*)(void*)g,
        (__attribute__((address_space(3))) i32x4*)(void*)l,
        0, 0);
#else
    *(uint4*)l = *(const uint4*)g;
#endif
}

__device__ __forceinline__ void cp16_wait() {
#if USE_ASYNC_LDS
#  if __has_builtin(__builtin_amdgcn_s_wait_asynccnt)
    __builtin_amdgcn_s_wait_asynccnt(0);
#  else
    asm volatile("s_wait_asynccnt 0x0" ::: "memory");
#  endif
#endif
}

__device__ __forceinline__ unsigned short f2bf(float f) {
    unsigned u = __builtin_bit_cast(unsigned, f);
    unsigned r = u + 0x7FFFu + ((u >> 16) & 1u);   // round-to-nearest-even
    return (unsigned short)(r >> 16);
}

#define WMMA_BF16(A, Bf, C) \
    __builtin_amdgcn_wmma_f32_16x16x32_bf16(false, (A), false, (Bf), (short)0, (C), false, false)

// ---------------------------------------------------------------------------
// Convert fp32 inputs [B][T][V] -> bf16 (4 elems / thread, 8B packed stores)
// ---------------------------------------------------------------------------
__global__ void k_cvt4(const float* __restrict__ src,
                       unsigned short* __restrict__ dst, size_t total4) {
    size_t i = (size_t)blockIdx.x * blockDim.x + threadIdx.x;
    if (i >= total4) return;
    const float4 f = ((const float4*)src)[i];
    unsigned long long pk =  (unsigned long long)f2bf(f.x)
                          | ((unsigned long long)f2bf(f.y) << 16)
                          | ((unsigned long long)f2bf(f.z) << 32)
                          | ((unsigned long long)f2bf(f.w) << 48);
    ((unsigned long long*)dst)[i] = pk;
}

// ---------------------------------------------------------------------------
// Transpose-convert: W [K][N] fp32 -> WT [N][K] bf16
// ---------------------------------------------------------------------------
__global__ void k_transpose_bf(const float* __restrict__ src,
                               unsigned short* __restrict__ dst,
                               int K, int N, size_t total) {
    size_t i = (size_t)blockIdx.x * blockDim.x + threadIdx.x;
    if (i >= total) return;
    size_t k = i / N, n = i % N;
    dst[n * (size_t)K + k] = f2bf(src[i]);
}

// ---------------------------------------------------------------------------
// Init: hs_bf slot 0 = bf16(h0), c workspace = c0
// ---------------------------------------------------------------------------
__global__ void k_init(const float* __restrict__ c0, const float* __restrict__ h0,
                       float* __restrict__ cws, unsigned short* __restrict__ hsbf) {
    int i = blockIdx.x * blockDim.x + threadIdx.x;
    if (i >= BB * HH) return;
    cws[i]  = c0[i];
    hsbf[i] = f2bf(h0[i]);
}

// ---------------------------------------------------------------------------
// One LSTM timestep, fully fused:
//   z = [x_t ; h_{t-1}] @ [WiT ; WhT]^T + b   (logical K = 2048, bf16 WMMA)
// Double-buffered LDS, async global->LDS staging, K-chunk = 64 per stage.
// Grid: 32 blocks (each owns 32 hidden units x all 4 gates), 256 threads.
// ---------------------------------------------------------------------------
__global__ void __launch_bounds__(256)
k_step(int t,
       const unsigned short* __restrict__ xbf,   // [B][T][V] bf16
       const unsigned short* __restrict__ WiT,   // [4H][V]   bf16
       const unsigned short* __restrict__ WhT,   // [4H][H]   bf16
       unsigned short* __restrict__ hsbf,        // [(T+1)][B][H] bf16
       float* __restrict__ cws,                  // [B][H] fp32
       float* __restrict__ hfws,                 // [B][H] fp32 (last h)
       const float* __restrict__ bias)           // [4H]
{
    __shared__ __attribute__((aligned(16))) unsigned short Abuf[2][32 * 64];
    __shared__ __attribute__((aligned(16))) unsigned short Bbuf[2][128 * 64];
    __shared__ float Zbuf[4 * 32 * 32];

    const int tid  = threadIdx.x;
    const int wave = tid >> 5;
    const int lane = tid & 31;
    const int half = lane >> 4;
    const int ln   = lane & 15;
    const int j0   = blockIdx.x * 32;   // hidden-unit chunk

    const int g  = wave >> 1;           // gate 0..3
    const int mt = wave & 1;            // batch tile 0..1

    // --- per-thread staging addresses, hoisted (no per-iter selects) -------
    // A: 32 rows x 64 K = 256 x 16B chunks, 1 per thread
    const int am = tid >> 3, aoff = (tid & 7) * 8;
    const unsigned short* pA[2] = {
        xbf  + (size_t)am * (TT * VV) + (size_t)t * VV + aoff,   // phase 0: x_t
        hsbf + ((size_t)t * BB + am) * HH + aoff                 // phase 1: h_{t-1}
    };
    const int adst = am * 64 + aoff;

    // B: 128 rows (4 gates x 32 cols) x 64 K = 1024 chunks, 4 per thread
    const unsigned short* pB[2][4];
    int bdst[4];
    #pragma unroll
    for (int s2 = 0; s2 < 4; ++s2) {
        int c = tid + s2 * 256;
        int r = c >> 3, off = (c & 7) * 8;
        int n = (r >> 5) * HH + j0 + (r & 31);
        pB[0][s2] = WiT + (size_t)n * VV + off;
        pB[1][s2] = WhT + (size_t)n * HH + off;
        bdst[s2]  = r * 64 + off;
    }

    auto issue = [&](int s, int buf) {
        const int ph = s >> 4, kb = (s & 15) * 64;
        cp16(pA[ph] + kb, &Abuf[buf][adst]);
        #pragma unroll
        for (int s2 = 0; s2 < 4; ++s2)
            cp16(pB[ph][s2] + kb, &Bbuf[buf][bdst[s2]]);
    };

    v8f acc0 = {}, acc1 = {};

    issue(0, 0);
    for (int s = 0; s < 32; ++s) {      // 32 stages x K=64 (2 phases of 1024)
        const int cur = s & 1;
        cp16_wait();
        __syncthreads();                 // stage s resident for all waves
        if (s + 1 < 32) issue(s + 1, cur ^ 1);

        const unsigned* A32 = (const unsigned*)Abuf[cur];
        const unsigned* B32 = (const unsigned*)Bbuf[cur];
        #pragma unroll
        for (int ks = 0; ks < 2; ++ks) { // two K=32 WMMA sub-steps
            Frag16 a, b0, b1;
            #pragma unroll
            for (int r = 0; r < 8; ++r) {
                int kd = (r < 4 ? r : r + 4) + half * 4;
                a.u[r] = A32[(mt * 16 + ln) * 32 + ks * 16 + kd];
            }
            #pragma unroll
            for (int r = 0; r < 8; ++r) {
                b0.u[r] = B32[(g * 32 + ln)      * 32 + ks * 16 + (r < 4 ? r : r + 4) + half * 4];
                b1.u[r] = B32[(g * 32 + 16 + ln) * 32 + ks * 16 + (r < 4 ? r : r + 4) + half * 4];
            }
            acc0 = WMMA_BF16(a.v, b0.v, acc0);
            acc1 = WMMA_BF16(a.v, b1.v, acc1);
        }
        __syncthreads();                 // all reads done before buffer reuse
    }

    // Epilogue: z + bias -> LDS, arranged [gate][batch][hidden]
    const float bia0 = bias[g * HH + j0 + ln];
    const float bia1 = bias[g * HH + j0 + 16 + ln];
    #pragma unroll
    for (int j = 0; j < 8; ++j) {
        int m = mt * 16 + j + half * 8;        // C/D layout: vgpr j, halves +8
        Zbuf[(g * 32 + m) * 32 + ln]      = acc0[j] + bia0;
        Zbuf[(g * 32 + m) * 32 + 16 + ln] = acc1[j] + bia1;
    }
    __syncthreads();

    // Fused LSTM cell update (1024 (b,j) pairs / 256 threads)
    #pragma unroll
    for (int p = 0; p < 4; ++p) {
        int idx = tid + p * 256;
        int bb = idx >> 5, jj = idx & 31;
        float zi = Zbuf[(0 * 32 + bb) * 32 + jj];
        float zf = Zbuf[(1 * 32 + bb) * 32 + jj];
        float zg = Zbuf[(2 * 32 + bb) * 32 + jj];
        float zo = Zbuf[(3 * 32 + bb) * 32 + jj];
        float iv = 1.f / (1.f + __expf(-zi));
        float fv = 1.f / (1.f + __expf(-zf));
        float gv = tanhf(zg);
        float ov = 1.f / (1.f + __expf(-zo));
        size_t ci = (size_t)bb * HH + j0 + jj;
        float cn = fv * cws[ci] + iv * gv;
        float hn = ov * tanhf(cn);
        cws[ci]  = cn;
        hfws[ci] = hn;
        hsbf[((size_t)(t + 1) * BB + bb) * HH + j0 + jj] = f2bf(hn);
    }
}

// ---------------------------------------------------------------------------
// logits[b*T+t][v] = hs[t+1][b][:] @ WdT[v][:] + bd[v]    (bf16 WMMA)
// 128x64 macro-tile, double-buffered K=64 stages, async staging.
// ---------------------------------------------------------------------------
__global__ void __launch_bounds__(256)
k_logits(const unsigned short* __restrict__ hsbf,   // [(T+1)][B][H] bf16
         const unsigned short* __restrict__ WdT,    // [V][H] bf16
         const float* __restrict__ bd,              // [V]
         float* __restrict__ out)                   // [B*T][V] fp32
{
    __shared__ __attribute__((aligned(16))) unsigned short Abuf[2][128 * 64];
    __shared__ __attribute__((aligned(16))) unsigned short Bbuf[2][64 * 64];

    const int tid  = threadIdx.x;
    const int wave = tid >> 5;
    const int lane = tid & 31;
    const int half = lane >> 4;
    const int ln   = lane & 15;
    const int bm = blockIdx.x * 128, bn = blockIdx.y * 64;
    const int mw = (wave & 3) * 32,  nw = (wave >> 2) * 32;

    // A: 128 rows x 64 K = 1024 chunks, 4/thread (rm = b*T+t -> hs[t+1][b])
    const unsigned short* pA[4];
    int adst[4];
    #pragma unroll
    for (int s2 = 0; s2 < 4; ++s2) {
        int c = tid + s2 * 256;
        int lr = c >> 3, off = (c & 7) * 8;
        int rm = bm + lr, b = rm >> 9, tq = rm & 511;
        pA[s2]  = hsbf + ((size_t)(tq + 1) * BB + b) * HH + off;
        adst[s2] = lr * 64 + off;
    }
    // B: 64 rows x 64 K = 512 chunks, 2/thread
    const unsigned short* pBc[2];
    int bdst[2];
    #pragma unroll
    for (int s2 = 0; s2 < 2; ++s2) {
        int c = tid + s2 * 256;
        int lr = c >> 3, off = (c & 7) * 8;
        pBc[s2] = WdT + (size_t)(bn + lr) * HH + off;
        bdst[s2] = lr * 64 + off;
    }

    auto issue = [&](int s, int buf) {
        const int kb = s * 64;
        #pragma unroll
        for (int s2 = 0; s2 < 4; ++s2) cp16(pA[s2] + kb, &Abuf[buf][adst[s2]]);
        #pragma unroll
        for (int s2 = 0; s2 < 2; ++s2) cp16(pBc[s2] + kb, &Bbuf[buf][bdst[s2]]);
    };

    v8f acc[2][2] = {};

    issue(0, 0);
    for (int s = 0; s < 16; ++s) {      // K = 1024 in stages of 64
        const int cur = s & 1;
        cp16_wait();
        __syncthreads();
        if (s + 1 < 16) issue(s + 1, cur ^ 1);

        const unsigned* A32 = (const unsigned*)Abuf[cur];
        const unsigned* B32 = (const unsigned*)Bbuf[cur];
        #pragma unroll
        for (int ks = 0; ks < 2; ++ks) {
            Frag16 a[2], b[2];
            #pragma unroll
            for (int mt2 = 0; mt2 < 2; ++mt2)
                #pragma unroll
                for (int r = 0; r < 8; ++r) {
                    int kd = (r < 4 ? r : r + 4) + half * 4;
                    a[mt2].u[r] = A32[(mw + mt2 * 16 + ln) * 32 + ks * 16 + kd];
                }
            #pragma unroll
            for (int nt2 = 0; nt2 < 2; ++nt2)
                #pragma unroll
                for (int r = 0; r < 8; ++r)
                    b[nt2].u[r] = B32[(nw + nt2 * 16 + ln) * 32 + ks * 16 + (r < 4 ? r : r + 4) + half * 4];
            #pragma unroll
            for (int mt2 = 0; mt2 < 2; ++mt2)
                #pragma unroll
                for (int nt2 = 0; nt2 < 2; ++nt2)
                    acc[mt2][nt2] = WMMA_BF16(a[mt2].v, b[nt2].v, acc[mt2][nt2]);
        }
        __syncthreads();
    }

    #pragma unroll
    for (int mt2 = 0; mt2 < 2; ++mt2)
        #pragma unroll
        for (int nt2 = 0; nt2 < 2; ++nt2) {
            int col = bn + nw + nt2 * 16 + ln;
            float bv = bd[col];
            #pragma unroll
            for (int j = 0; j < 8; ++j) {
                size_t rm = (size_t)bm + mw + mt2 * 16 + j + half * 8;
                out[rm * VV + col] = acc[mt2][nt2][j] + bv;
            }
        }
}

// ---------------------------------------------------------------------------
// In-place log-softmax over V=1024 per row (one block per row)
// ---------------------------------------------------------------------------
__global__ void __launch_bounds__(256)
k_logsoftmax(float* __restrict__ logp) {
    __shared__ float red[256];
    const int tid = threadIdx.x;
    float* p = logp + (size_t)blockIdx.x * VV;

    float v[4], mx = -INFINITY;
    #pragma unroll
    for (int i = 0; i < 4; ++i) { v[i] = p[tid + i * 256]; mx = fmaxf(mx, v[i]); }
    red[tid] = mx; __syncthreads();
    for (int s = 128; s > 0; s >>= 1) {
        if (tid < s) red[tid] = fmaxf(red[tid], red[tid + s]);
        __syncthreads();
    }
    mx = red[0]; __syncthreads();

    float sm = 0.f;
    #pragma unroll
    for (int i = 0; i < 4; ++i) sm += __expf(v[i] - mx);
    red[tid] = sm; __syncthreads();
    for (int s = 128; s > 0; s >>= 1) {
        if (tid < s) red[tid] += red[tid + s];
        __syncthreads();
    }
    const float lz = mx + __logf(red[0]);
    #pragma unroll
    for (int i = 0; i < 4; ++i) p[tid + i * 256] = v[i] - lz;
}

// ---------------------------------------------------------------------------
// Final: d_out[0:32768] = c_f, d_out[32768:65536] = h_f
// ---------------------------------------------------------------------------
__global__ void k_final(const float* __restrict__ cws,
                        const float* __restrict__ hfws,
                        float* __restrict__ out) {
    int i = blockIdx.x * blockDim.x + threadIdx.x;
    if (i >= 2 * BB * HH) return;
    out[i] = (i < BB * HH) ? cws[i] : hfws[i - BB * HH];
}

// ---------------------------------------------------------------------------
extern "C" void kernel_launch(void* const* d_in, const int* in_sizes, int n_in,
                              void* d_out, int out_size, void* d_ws, size_t ws_size,
                              hipStream_t stream) {
    const float* c0     = (const float*)d_in[0];
    const float* h0     = (const float*)d_in[1];
    const float* inputs = (const float*)d_in[2];
    const float* Wi     = (const float*)d_in[3];
    const float* Wh     = (const float*)d_in[4];
    const float* b      = (const float*)d_in[5];
    const float* Wd     = (const float*)d_in[6];
    const float* bd     = (const float*)d_in[7];
    float* out = (float*)d_out;

    // Workspace carve (all 256B aligned): ~86 MB total
    char* w = (char*)d_ws;
    size_t off = 0;
    auto carve = [&](size_t bytes) -> void* {
        void* p = w + off;
        off += (bytes + 255) & ~(size_t)255;
        return p;
    };
    unsigned short* xbf  = (unsigned short*)carve((size_t)BB * TT * VV * 2);
    unsigned short* WiT  = (unsigned short*)carve((size_t)G4 * VV * 2);
    unsigned short* WhT  = (unsigned short*)carve((size_t)G4 * HH * 2);
    unsigned short* WdT  = (unsigned short*)carve((size_t)VV * HH * 2);
    unsigned short* hsbf = (unsigned short*)carve((size_t)(TT + 1) * BB * HH * 2);
    float* cws  = (float*)carve((size_t)BB * HH * 4);
    float* hfws = (float*)carve((size_t)BB * HH * 4);
    (void)ws_size; (void)in_sizes; (void)n_in; (void)out_size;

    // 1) Precision conversion + weight transposes (bf16, [N][K])
    {
        size_t tot4 = (size_t)BB * TT * VV / 4;                 // 4,194,304
        k_cvt4<<<(unsigned)((tot4 + 255) / 256), 256, 0, stream>>>(inputs, xbf, tot4);
    }
    {
        size_t tot = (size_t)VV * G4;                           // 4,194,304
        k_transpose_bf<<<(unsigned)((tot + 255) / 256), 256, 0, stream>>>(Wi, WiT, VV, G4, tot);
        k_transpose_bf<<<(unsigned)((tot + 255) / 256), 256, 0, stream>>>(Wh, WhT, HH, G4, tot);
    }
    {
        size_t tot = (size_t)HH * VV;                           // 1,048,576
        k_transpose_bf<<<(unsigned)((tot + 255) / 256), 256, 0, stream>>>(Wd, WdT, HH, VV, tot);
    }
    k_init<<<(BB * HH + 255) / 256, 256, 0, stream>>>(c0, h0, cws, hsbf);

    // 2) Sequential LSTM scan: one fused GEMM+cell kernel per timestep
    for (int t = 0; t < TT; ++t)
        k_step<<<HH / 32, 256, 0, stream>>>(t, xbf, WiT, WhT, hsbf, cws, hfws, b);

    // 3) Decoder GEMM -> logits straight into d_out, then in-place log-softmax
    float* logp = out + 2 * BB * HH;
    k_logits<<<dim3((BB * TT) / 128, VV / 64), 256, 0, stream>>>(hsbf, WdT, bd, logp);
    k_logsoftmax<<<BB * TT, 256, 0, stream>>>(logp);

    // 4) Final states
    k_final<<<(2 * BB * HH + 255) / 256, 256, 0, stream>>>(cws, hfws, out);
}
